// Remamba_68247030333812
// MI455X (gfx1250) — compile-verified, matrix operand order checked by reading, not run on hardware
//
#include <hip/hip_runtime.h>
#include <math.h>

typedef __attribute__((ext_vector_type(16))) _Float16 v16h;
typedef __attribute__((ext_vector_type(8)))  float    v8f;

__device__ __forceinline__ float silu_f(float v) { return v / (1.0f + __expf(-v)); }

// ---------------------------------------------------------------------------
// Register-blocked wave32 WMMA GEMM:
//   C[m][n] = scale * sum_k A[m][k]*B'[n][k] + bias[n]
//   BT=false : B stored row-major (N x K)  ->  C = A @ B^T
//   BT=true  : B stored row-major (K x N)  ->  C = A @ B
// One wave computes a 32(M) x 64(N) tile: 2 A-fragments x 4 B-fragments ->
// 8 accumulators, 8 x v_wmma_f32_16x16x32_f16 per K-step. A reused 4x, B 2x.
// Requires K % 32 == 0, M % 32 == 0, N % 64 == 0 (true for all GEMMs here).
// Batched via blockIdx.z with element strides sAb/sBb/sCb.
// ---------------------------------------------------------------------------
template <bool BT>
__global__ __launch_bounds__(32)
void wmma_gemm_f16(const float* __restrict__ A, long sAb, int lda,
                   const float* __restrict__ Bm, long sBb, int ldb,
                   float* __restrict__ C, long sCb, int ldc,
                   const float* __restrict__ bias, float scale, int K)
{
    const int lane = threadIdx.x;
    const int hi   = lane >> 4;    // K half-select per ISA fragment layout
    const int li   = lane & 15;
    A  += (long)blockIdx.z * sAb;
    Bm += (long)blockIdx.z * sBb;
    C  += (long)blockIdx.z * sCb;
    const int m0 = blockIdx.x * 32;
    const int n0 = blockIdx.y * 64;

    v8f acc[2][4] = {};

    for (int k0 = 0; k0 < K; k0 += 32) {
        v16h a[2], b[4];
        // A fragments: lane holds row m0+16*mi+li; elems 0..7 -> k0+8*hi+j,
        // elems 8..15 -> k0+16+8*hi+j  (16x32 f16 A layout, wave32)
#pragma unroll
        for (int mi = 0; mi < 2; ++mi) {
            const float* ap = A + (long)(m0 + 16 * mi + li) * lda + k0 + 8 * hi;
            __builtin_prefetch(ap + 32, 0, 1);
#pragma unroll
            for (int j = 0; j < 8; ++j) {
                a[mi][j]     = (_Float16)ap[j];
                a[mi][j + 8] = (_Float16)ap[j + 16];
            }
        }
#pragma unroll
        for (int nt = 0; nt < 4; ++nt) {
            if (BT) {   // B stored KxN: column n0+16*nt+li, strided in k
                const float* bp = Bm + (long)(k0 + 8 * hi) * ldb + n0 + 16 * nt + li;
#pragma unroll
                for (int j = 0; j < 8; ++j) {
                    b[nt][j]     = (_Float16)bp[(long)j * ldb];
                    b[nt][j + 8] = (_Float16)bp[(long)(j + 16) * ldb];
                }
            } else {    // B stored NxK: row n0+16*nt+li, contiguous in k
                const float* bp = Bm + (long)(n0 + 16 * nt + li) * ldb + k0 + 8 * hi;
#pragma unroll
                for (int j = 0; j < 8; ++j) {
                    b[nt][j]     = (_Float16)bp[j];
                    b[nt][j + 8] = (_Float16)bp[j + 16];
                }
            }
        }
#pragma unroll
        for (int mi = 0; mi < 2; ++mi)
#pragma unroll
            for (int nt = 0; nt < 4; ++nt)
                acc[mi][nt] = __builtin_amdgcn_wmma_f32_16x16x32_f16(
                    false, a[mi], false, b[nt], (short)0, acc[mi][nt], false, false);
    }

    // C/D layout: lane -> N = li, VGPR r -> M = r + 8*hi
#pragma unroll
    for (int nt = 0; nt < 4; ++nt) {
        const int   col = n0 + 16 * nt + li;
        const float bv  = bias ? bias[col] : 0.0f;
#pragma unroll
        for (int mi = 0; mi < 2; ++mi) {
            const int rbase = m0 + 16 * mi + 8 * hi;
#pragma unroll
            for (int r = 0; r < 8; ++r)
                C[(long)(rbase + r) * ldc + col] = acc[mi][nt][r] * scale + bv;
        }
    }
}

// ---------------------------------------------------------------------------
// Depthwise causal conv (width 4) + SiLU.  XZ row stride 2048 (xm = cols 0..1023)
// ---------------------------------------------------------------------------
__global__ __launch_bounds__(256)
void conv_silu_k(const float* __restrict__ XZ, const float* __restrict__ cw,
                 const float* __restrict__ cb, float* __restrict__ XC)
{
    int idx = blockIdx.x * 256 + threadIdx.x;   // over 2048*1024
    int d  = idx & 1023;
    int bl = idx >> 10;
    int l  = bl & 511;
    float acc = cb[d];
#pragma unroll
    for (int k = 0; k < 4; ++k) {
        int ll = l - 3 + k;
        if (ll >= 0)
            acc += XZ[(long)(bl - 3 + k) * 2048 + d] * cw[d * 4 + k];
    }
    XC[(long)bl * 1024 + d] = silu_f(acc);
}

__global__ __launch_bounds__(256)
void softplus_ip_k(float* __restrict__ DT)
{
    long i  = (long)blockIdx.x * 256 + threadIdx.x;
    float x = DT[i];
    DT[i]   = (x > 20.0f) ? x : log1pf(__expf(x));
}

// ---------------------------------------------------------------------------
// Chunked (parallel) selective scan over L=512 in NC=8 chunks of 64.
// Recurrence h_t = dA_t*h + dBu_t is associative; per chunk we compute the
// decay product P and zero-start end state E, combine sequentially over the
// 8 chunks (tiny), then re-scan each chunk from its incoming state H0.
// Layouts keyed as [((b*8+c)*16+s)*1024 + d] so all accesses coalesce in d.
// ---------------------------------------------------------------------------
__global__ __launch_bounds__(256)
void scanA_k(const float* __restrict__ XC, const float* __restrict__ DBC,
             const float* __restrict__ DT, const float* __restrict__ A_log,
             float* __restrict__ P, float* __restrict__ E)
{
    int tid = blockIdx.x * 256 + threadIdx.x;   // 0..32767
    int d  = tid & 1023;
    int bc = tid >> 10;        // b*8 + c
    int b  = bc >> 3;
    int c  = bc & 7;
    float An[16], Pr[16], Er[16];
#pragma unroll
    for (int s = 0; s < 16; ++s) {
        An[s] = -__expf(A_log[d * 16 + s]);
        Pr[s] = 1.0f; Er[s] = 0.0f;
    }
    const int l0 = c * 64;
    for (int l = l0; l < l0 + 64; ++l) {
        long r   = (long)b * 512 + l;
        float dt = DT[r * 1024 + d];
        float xc = XC[r * 1024 + d];
        const float* dbc = DBC + r * 64;
#pragma unroll
        for (int s = 0; s < 16; ++s) {
            float dA = __expf(dt * An[s]);
            Pr[s] *= dA;
            Er[s] = fmaf(dA, Er[s], dt * dbc[32 + s] * xc);
        }
    }
    long base = ((long)bc * 16) * 1024 + d;
#pragma unroll
    for (int s = 0; s < 16; ++s) {
        P[base + (long)s * 1024] = Pr[s];
        E[base + (long)s * 1024] = Er[s];
    }
}

__global__ __launch_bounds__(256)
void scanB_k(const float* __restrict__ P, const float* __restrict__ E,
             float* __restrict__ H0)
{
    int tid = blockIdx.x * 256 + threadIdx.x;   // 0..4095
    int d = tid & 1023;
    int b = tid >> 10;
    float h[16];
#pragma unroll
    for (int s = 0; s < 16; ++s) h[s] = 0.0f;
    for (int c = 0; c < 8; ++c) {
        long base = ((long)(b * 8 + c) * 16) * 1024 + d;
#pragma unroll
        for (int s = 0; s < 16; ++s) {
            long i = base + (long)s * 1024;
            H0[i] = h[s];
            h[s]  = fmaf(P[i], h[s], E[i]);
        }
    }
}

__global__ __launch_bounds__(256)
void scanC_k(const float* __restrict__ XZ, const float* __restrict__ XC,
             const float* __restrict__ DBC, const float* __restrict__ DT,
             const float* __restrict__ A_log, const float* __restrict__ Dskip,
             const float* __restrict__ H0, float* __restrict__ YZ)
{
    int tid = blockIdx.x * 256 + threadIdx.x;   // 0..32767
    int d  = tid & 1023;
    int bc = tid >> 10;
    int b  = bc >> 3;
    int c  = bc & 7;
    float An[16], h[16];
    long base = ((long)bc * 16) * 1024 + d;
#pragma unroll
    for (int s = 0; s < 16; ++s) {
        An[s] = -__expf(A_log[d * 16 + s]);
        h[s]  = H0[base + (long)s * 1024];
    }
    const float dsk = Dskip[d];
    const int l0 = c * 64;
    for (int l = l0; l < l0 + 64; ++l) {
        long r   = (long)b * 512 + l;
        float dt = DT[r * 1024 + d];
        float xc = XC[r * 1024 + d];
        const float* dbc = DBC + r * 64;   // [32:48)=B, [48:64)=C
        float y = 0.0f;
#pragma unroll
        for (int s = 0; s < 16; ++s) {
            float dA = __expf(dt * An[s]);
            h[s] = fmaf(dA, h[s], dt * dbc[32 + s] * xc);
            y    = fmaf(h[s], dbc[48 + s], y);
        }
        y = fmaf(xc, dsk, y);
        float z = XZ[r * 2048 + 1024 + d];
        YZ[r * 1024 + d] = y * silu_f(z);
    }
}

// ---------------------------------------------------------------------------
// Row softmax over 256 columns (scale folded into the score GEMM)
// ---------------------------------------------------------------------------
__global__ __launch_bounds__(256)
void softmax_rows_k(float* __restrict__ S)
{
    __shared__ float red[256];
    float* row = S + (long)blockIdx.x * 256;
    int t = threadIdx.x;
    float v = row[t];
    red[t] = v; __syncthreads();
    for (int s = 128; s > 0; s >>= 1) { if (t < s) red[t] = fmaxf(red[t], red[t + s]); __syncthreads(); }
    float mx = red[0]; __syncthreads();
    float e = __expf(v - mx);
    red[t] = e; __syncthreads();
    for (int s = 128; s > 0; s >>= 1) { if (t < s) red[t] += red[t + s]; __syncthreads(); }
    row[t] = e * (1.0f / red[0]);
}

// ---------------------------------------------------------------------------
// r = m + attn; LayerNorm over 512; 256 threads x 2 elems per row
// ---------------------------------------------------------------------------
__global__ __launch_bounds__(256)
void residual_ln_k(const float* __restrict__ Mb, const float* __restrict__ AO,
                   const float* __restrict__ g, const float* __restrict__ be,
                   float* __restrict__ out)
{
    __shared__ float red[256];
    long base = (long)blockIdx.x * 512;
    int t = threadIdx.x;
    float x0 = Mb[base + t]       + AO[base + t];
    float x1 = Mb[base + t + 256] + AO[base + t + 256];
    red[t] = x0 + x1; __syncthreads();
    for (int s = 128; s > 0; s >>= 1) { if (t < s) red[t] += red[t + s]; __syncthreads(); }
    float mu = red[0] * (1.0f / 512.0f); __syncthreads();
    float d0 = x0 - mu, d1 = x1 - mu;
    red[t] = d0 * d0 + d1 * d1; __syncthreads();
    for (int s = 128; s > 0; s >>= 1) { if (t < s) red[t] += red[t + s]; __syncthreads(); }
    float rstd = rsqrtf(red[0] * (1.0f / 512.0f) + 1e-5f);
    out[base + t]       = d0 * rstd * g[t]       + be[t];
    out[base + t + 256] = d1 * rstd * g[t + 256] + be[t + 256];
}

// ---------------------------------------------------------------------------
extern "C" void kernel_launch(void* const* d_in, const int* in_sizes, int n_in,
                              void* d_out, int out_size, void* d_ws, size_t ws_size,
                              hipStream_t stream)
{
    const float* x         = (const float*)d_in[0];   // (4,512,512)
    const float* in_proj_w = (const float*)d_in[1];   // (2048,512)
    const float* conv_w    = (const float*)d_in[2];   // (1024,4)
    const float* conv_b    = (const float*)d_in[3];   // (1024)
    const float* x_proj_w  = (const float*)d_in[4];   // (64,1024)
    const float* dt_proj_w = (const float*)d_in[5];   // (1024,32)
    const float* dt_proj_b = (const float*)d_in[6];   // (1024)
    const float* A_log     = (const float*)d_in[7];   // (1024,16)
    const float* D_skip    = (const float*)d_in[8];   // (1024)
    const float* out_proj_w= (const float*)d_in[9];   // (512,1024)
    const float* attn_in_w = (const float*)d_in[10];  // (1536,512)
    const float* attn_in_b = (const float*)d_in[11];  // (1536)
    const float* attn_out_w= (const float*)d_in[12];  // (512,512)
    const float* attn_out_b= (const float*)d_in[13];  // (512)
    const float* ln_g      = (const float*)d_in[14];  // (512)
    const float* ln_b      = (const float*)d_in[15];  // (512)
    float* out = (float*)d_out;

    // Workspace layout (floats), with aliasing of dead regions:
    float* ws  = (float*)d_ws;
    float* XZ  = ws;                       // 2048 x 2048   (dead after scanC)
    float* XC  = XZ  + 2048L * 2048;       // 2048 x 1024   (dead after scanC)
    float* DBC = XC  + 2048L * 1024;       // 2048 x 64
    float* DT  = DBC + 2048L * 64;         // 2048 x 1024   (dead after scanC)
    float* YZ  = DT  + 2048L * 1024;       // 2048 x 1024
    float* Mb  = YZ  + 2048L * 1024;       // 2048 x 512
    float* Sb  = Mb  + 2048L * 512;        // 8 x 256 x 256 scores
    float* Pb  = Sb  + 8L * 256 * 256;     // 4x8x16x1024 chunk decay products
    float* Eb  = Pb  + 4L * 8 * 16 * 1024; // 4x8x16x1024 chunk end states
    float* H0b = Eb  + 4L * 8 * 16 * 1024; // 4x8x16x1024 chunk incoming states
    float* QKV = XZ;                       // 2048 x 1536  (aliases XZ)
    float* Ob  = XC;                       // 2048 x 512   (aliases XC)
    float* AO  = DT;                       // 2048 x 512   (aliases DT)

    // 1) xz = x @ in_proj_w^T        (M=2048,N=2048,K=512)
    wmma_gemm_f16<false><<<dim3(64, 32, 1), 32, 0, stream>>>(
        x, 0, 512, in_proj_w, 0, 512, XZ, 0, 2048, nullptr, 1.0f, 512);

    // 2) causal conv4 + SiLU -> XC
    conv_silu_k<<<(2048 * 1024) / 256, 256, 0, stream>>>(XZ, conv_w, conv_b, XC);

    // 3) dbc = xc @ x_proj_w^T       (M=2048,N=64,K=1024)
    wmma_gemm_f16<false><<<dim3(64, 1, 1), 32, 0, stream>>>(
        XC, 0, 1024, x_proj_w, 0, 1024, DBC, 0, 64, nullptr, 1.0f, 1024);

    // 4) dt = dbc[:, :32] @ dt_proj_w^T + dt_proj_b  (M=2048,N=1024,K=32)
    wmma_gemm_f16<false><<<dim3(64, 16, 1), 32, 0, stream>>>(
        DBC, 0, 64, dt_proj_w, 0, 32, DT, 0, 1024, dt_proj_b, 1.0f, 32);
    softplus_ip_k<<<(2048 * 1024) / 256, 256, 0, stream>>>(DT);

    // 5) chunked selective scan (8 chunks of 64) + D_skip + silu(z) gate -> YZ
    scanA_k<<<128, 256, 0, stream>>>(XC, DBC, DT, A_log, Pb, Eb);
    scanB_k<<<16, 256, 0, stream>>>(Pb, Eb, H0b);
    scanC_k<<<128, 256, 0, stream>>>(XZ, XC, DBC, DT, A_log, D_skip, H0b, YZ);

    // 6) m = YZ @ out_proj_w^T       (M=2048,N=512,K=1024)
    wmma_gemm_f16<false><<<dim3(64, 8, 1), 32, 0, stream>>>(
        YZ, 0, 1024, out_proj_w, 0, 1024, Mb, 0, 512, nullptr, 1.0f, 1024);

    // 7) qkv = m @ attn_in_w^T + attn_in_b   (M=2048,N=1536,K=512)
    wmma_gemm_f16<false><<<dim3(64, 24, 1), 32, 0, stream>>>(
        Mb, 0, 512, attn_in_w, 0, 512, QKV, 0, 1536, attn_in_b, 1.0f, 512);

    // 8) scores = q @ k^T / sqrt(512)  per chunk  (8 x [256 x 256], K=512)
    const float iscale = 1.0f / sqrtf(512.0f);
    wmma_gemm_f16<false><<<dim3(8, 4, 8), 32, 0, stream>>>(
        QKV, 256L * 1536, 1536, QKV + 512, 256L * 1536, 1536,
        Sb, 256L * 256, 256, nullptr, iscale, 512);

    // 9) softmax rows
    softmax_rows_k<<<8 * 256, 256, 0, stream>>>(Sb);

    // 10) o = p @ v   per chunk (B is KxN -> BT variant), M=256,N=512,K=256
    wmma_gemm_f16<true><<<dim3(8, 8, 8), 32, 0, stream>>>(
        Sb, 256L * 256, 256, QKV + 1024, 256L * 1536, 1536,
        Ob, 256L * 512, 512, nullptr, 1.0f, 256);

    // 11) attn = o @ attn_out_w^T + attn_out_b   (M=2048,N=512,K=512)
    wmma_gemm_f16<false><<<dim3(64, 8, 1), 32, 0, stream>>>(
        Ob, 0, 512, attn_out_w, 0, 512, AO, 0, 512, attn_out_b, 1.0f, 512);

    // 12) r = m + attn; LayerNorm -> out
    residual_ln_k<<<2048, 256, 0, stream>>>(Mb, AO, ln_g, ln_b, out);
}
